// SparseFeedForward_71476845740788
// MI455X (gfx1250) — matrix operand order repaired and checked by reading, not compiled, not run on hardware
//
#include <hip/hip_runtime.h>
#include <math.h>

// ---------------------------------------------------------------------------
// MoE SwiGLU top-2/8, D_MODEL=1024, D_FF=4096, 4096 tokens — MI455X (gfx1250).
//
//   1) moe_zero     : zero out + expert counters
//   2) moe_gate     : wave32-per-token gate GEMV, softmax, top-2, scatter
//   3) convert_w ×3 : fp32 weights -> bf16 WMMA-tile-major (1KB tiles);
//                     per-expert bf16 set (24 MB) stays in the 192 MB L2.
//   4) moe_ffn      : fused SwiGLU FFN per (expert, 16-token tile).
//                     X rows gathered by the Tensor Data Mover (gather D#),
//                     B tiles as 2x global_load_b128 per WMMA, software-
//                     pipelined one tile deep so waits are partial instead
//                     of drain-to-zero.
// ---------------------------------------------------------------------------

#define D_MODEL 1024
#define D_FF    4096
#define NUM_E   8
#define N_TOK   4096
#define CAP_E   4096

typedef __attribute__((ext_vector_type(16))) __bf16 v16bf;
typedef __attribute__((ext_vector_type(8)))  float  v8f;
typedef __attribute__((ext_vector_type(4)))  unsigned int v4u;
typedef __attribute__((ext_vector_type(8)))  int v8i;
typedef __attribute__((ext_vector_type(4)))  int v4i;

struct B256 { uint4 lo, hi; };   // 32 bytes == one v16bf

// fp32 -> bf16 RNE, result in [31:16]
__device__ __forceinline__ unsigned int bfrnd(float f) {
  unsigned int u = __float_as_uint(f);
  return u + 0x7FFFu + ((u >> 16) & 1u);
}
// two fp32 -> packed bf16 pair via v_perm_b32
__device__ __forceinline__ unsigned int pack2(float a, float b) {
  return __builtin_amdgcn_perm(bfrnd(b), bfrnd(a), 0x07060302u);
}
__device__ __forceinline__ unsigned short f2bf(float f) {
  return (unsigned short)(bfrnd(f) >> 16);
}

__device__ __forceinline__ B256 load2_raw(const unsigned short* p0,
                                          const unsigned short* p1) {
  B256 t;
  t.lo = *(const uint4*)p0;
  t.hi = *(const uint4*)p1;
  return t;
}
__device__ __forceinline__ v16bf as_bf(B256 t) {
  return __builtin_bit_cast(v16bf, t);
}

// ---------------------------------------------------------------------------
__global__ __launch_bounds__(256) void moe_zero(float* __restrict__ out, int n,
                                                int* __restrict__ cnt) {
  int idx = blockIdx.x * 256 + threadIdx.x;
  if (idx < n) out[idx] = 0.0f;
  if (idx < NUM_E) cnt[idx] = 0;
}

// ---------------------------------------------------------------------------
// Gate: one wave32 per token.
// ---------------------------------------------------------------------------
__global__ __launch_bounds__(256) void moe_gate(const float* __restrict__ x,
                                                const float* __restrict__ Wg,
                                                const float* __restrict__ bgv,
                                                int*  __restrict__ cnt,
                                                int*  __restrict__ tok,
                                                float* __restrict__ wl) {
  const int lane = threadIdx.x & 31;
  const int t = blockIdx.x * 8 + (threadIdx.x >> 5);
  const float* xr = x + (size_t)t * D_MODEL;

  float p[NUM_E];
#pragma unroll
  for (int e = 0; e < NUM_E; ++e) p[e] = 0.0f;

  for (int i = lane; i < D_MODEL; i += 32) {
    float xv = xr[i];
    const float4* wr = (const float4*)(Wg + i * NUM_E);
    float4 w0 = wr[0], w1 = wr[1];
    p[0] += xv * w0.x; p[1] += xv * w0.y; p[2] += xv * w0.z; p[3] += xv * w0.w;
    p[4] += xv * w1.x; p[5] += xv * w1.y; p[6] += xv * w1.z; p[7] += xv * w1.w;
  }
#pragma unroll
  for (int e = 0; e < NUM_E; ++e) {
    float v = p[e];
#pragma unroll
    for (int off = 16; off > 0; off >>= 1) v += __shfl_xor(v, off, 32);
    p[e] = v;
  }

  if (lane == 0) {
    float l[NUM_E];
    float mx = -3.4e38f;
#pragma unroll
    for (int e = 0; e < NUM_E; ++e) { l[e] = p[e] + bgv[e]; mx = fmaxf(mx, l[e]); }
    float sum = 0.0f;
#pragma unroll
    for (int e = 0; e < NUM_E; ++e) { l[e] = __expf(l[e] - mx); sum += l[e]; }
    float inv = 1.0f / sum;
#pragma unroll
    for (int e = 0; e < NUM_E; ++e) l[e] *= inv;

    int i0 = 0;
#pragma unroll
    for (int e = 1; e < NUM_E; ++e) if (l[e] > l[i0]) i0 = e;
    float v0 = l[i0]; l[i0] = -1.0f;
    int i1 = 0;
#pragma unroll
    for (int e = 1; e < NUM_E; ++e) if (l[e] > l[i1]) i1 = e;
    float v1 = l[i1];

    float norm = 1.0f / (v0 + v1 + 1e-6f);
    int pos0 = atomicAdd(&cnt[i0], 1);
    tok[i0 * CAP_E + pos0] = t;  wl[i0 * CAP_E + pos0] = v0 * norm;
    int pos1 = atomicAdd(&cnt[i1], 1);
    tok[i1 * CAP_E + pos1] = t;  wl[i1 * CAP_E + pos1] = v1 * norm;
  }
}

// ---------------------------------------------------------------------------
// Weight pre-conversion: fp32 [E][K][N] -> bf16 WMMA tiles (1KB each).
// ---------------------------------------------------------------------------
__global__ __launch_bounds__(256) void convert_w(const float* __restrict__ src,
                                                 unsigned short* __restrict__ dst,
                                                 int nkt, int nnt, int Ntot) {
  const int lane = threadIdx.x & 31;
  const int tile = blockIdx.x * 8 + (threadIdx.x >> 5);
  const int kt = tile % nkt;
  const int nt = (tile / nkt) % nnt;
  const int e  = tile / (nkt * nnt);

  const float* s = src + ((size_t)e * nkt * 32 + (size_t)kt * 32 + ((lane >> 4) << 3)) * Ntot
                       + nt * 16 + (lane & 15);
  const size_t N = (size_t)Ntot;
  float4 f0 = make_float4(s[0 * N], s[1 * N], s[2 * N], s[3 * N]);
  float4 f1 = make_float4(s[4 * N], s[5 * N], s[6 * N], s[7 * N]);
  float4 f2 = make_float4(s[16 * N], s[17 * N], s[18 * N], s[19 * N]);
  float4 f3 = make_float4(s[20 * N], s[21 * N], s[22 * N], s[23 * N]);

  uint4 o0, o1;
  o0.x = pack2(f0.x, f0.y); o0.y = pack2(f0.z, f0.w);
  o0.z = pack2(f1.x, f1.y); o0.w = pack2(f1.z, f1.w);
  o1.x = pack2(f2.x, f2.y); o1.y = pack2(f2.z, f2.w);
  o1.z = pack2(f3.x, f3.y); o1.w = pack2(f3.z, f3.w);

  unsigned short* d = dst + (size_t)tile * 512 + lane * 16;
  *(uint4*)d = o0;
  *(uint4*)(d + 8) = o1;
}

// ---------------------------------------------------------------------------
// Fused expert FFN.
// ---------------------------------------------------------------------------
__global__ __launch_bounds__(256) void moe_ffn(const float* __restrict__ x,
                                               const unsigned short* __restrict__ W1t,
                                               const float* __restrict__ b1v,
                                               const unsigned short* __restrict__ W3t,
                                               const float* __restrict__ b3v,
                                               const unsigned short* __restrict__ W2t,
                                               const float* __restrict__ b2v,
                                               const int*  __restrict__ cnt,
                                               const int*  __restrict__ tok,
                                               const float* __restrict__ wl,
                                               float* __restrict__ out) {
  __shared__ unsigned short Ash[16 * D_MODEL];  // bf16 A tile         (32 KB)
  __shared__ float          Xst[4 * D_MODEL];   // TDM gather staging  (16 KB)
  __shared__ unsigned short Hsh[16 * 128];      // bf16 H chunk tile   ( 4 KB)

  const int e    = blockIdx.x >> 8;
  const int row0 = (blockIdx.x & 255) << 4;
  int nrow = cnt[e] - row0;
  if (nrow <= 0) return;
  if (nrow > 16) nrow = 16;

  const int tid  = threadIdx.x;
  const int lane = tid & 31;
  const int wvid = tid >> 5;
  const int m    = lane & 15;
  const int half = lane >> 4;

  // ---- stage A: TDM gather-mode, 4 token rows per descriptor -------------
  for (int it = 0; it < 4; ++it) {
#if __has_builtin(__builtin_amdgcn_tensor_load_to_lds)
    if (wvid == 0) {
      int r0 = it * 4;
      unsigned t0 = (r0 + 0 < nrow) ? (unsigned)tok[e * CAP_E + row0 + r0 + 0] : 0u;
      unsigned t1 = (r0 + 1 < nrow) ? (unsigned)tok[e * CAP_E + row0 + r0 + 1] : 0u;
      unsigned t2 = (r0 + 2 < nrow) ? (unsigned)tok[e * CAP_E + row0 + r0 + 2] : 0u;
      unsigned t3 = (r0 + 3 < nrow) ? (unsigned)tok[e * CAP_E + row0 + r0 + 3] : 0u;

      unsigned long long ga = (unsigned long long)(uintptr_t)(const void*)x;
      v4u g0;
      g0.x = 0x80000001u;                         // count=1, gather_mode=1, idx16
      g0.y = (unsigned)(uintptr_t)(void*)Xst;     // LDS byte offset
      g0.z = (unsigned)(ga & 0xffffffffu);        // global_addr[31:0]
      g0.w = (unsigned)((ga >> 32) & 0x1ffffffu) | 0x80000000u;  // type=2
      v8i g1;
      g1[0] = 0x00020000;          // data_size = 4B, workgroup_mask = 0
      g1[1] = (int)(1024u << 16);  // tensor_dim0 = 1024 elems
      g1[2] = (int)(4096u << 16);  // tensor_dim1 = 4096 rows
      g1[3] = (int)(1024u << 16);  // tile_dim0 = 1024 elems (full row)
      g1[4] = 4;                   // tile_dim1 = 4 valid gather indices
      g1[5] = 1024;                // tensor_dim0_stride = 1024 elems
      g1[6] = 0;
      g1[7] = 0;
      v4i g2;
      g2[0] = (int)(t0 | (t1 << 16));
      g2[1] = (int)(t2 | (t3 << 16));
      g2[2] = 0; g2[3] = 0;
      v4i g3;
      g3[0] = 0; g3[1] = 0; g3[2] = 0; g3[3] = 0;
      v8i g4;   // extra operand of the 6-arg builtin variant (unused: zeros)
      g4[0] = 0; g4[1] = 0; g4[2] = 0; g4[3] = 0;
      g4[4] = 0; g4[5] = 0; g4[6] = 0; g4[7] = 0;
      __builtin_amdgcn_tensor_load_to_lds(g0, g1, g2, g3, g4, 0);
      __builtin_amdgcn_s_wait_tensorcnt(0);
    }
#else
    {   // fallback: cooperative vector loads
      int rloc = tid >> 6;
      int r = it * 4 + rloc;
      int c0 = (tid & 63) << 4;
      int trow = (r < nrow) ? tok[e * CAP_E + row0 + r] : 0;
      const float4* src = (const float4*)(x + (size_t)trow * D_MODEL + c0);
#pragma unroll
      for (int j = 0; j < 4; ++j) ((float4*)(Xst + rloc * D_MODEL + c0))[j] = src[j];
    }
#endif
    __syncthreads();
    {   // convert 4 rows fp32 -> bf16 into Ash (zero rows beyond nrow)
      int rloc = tid >> 6;                 // 0..3
      int r = it * 4 + rloc;
      int c0 = (tid & 63) << 4;            // 16 floats per thread
      const float4* sv = (const float4*)(Xst + rloc * D_MODEL + c0);
      float4 f0 = sv[0], f1 = sv[1], f2 = sv[2], f3 = sv[3];
      if (r >= nrow) {
        f0 = f1 = f2 = f3 = make_float4(0.f, 0.f, 0.f, 0.f);
      }
      uint4 o0, o1;
      o0.x = pack2(f0.x, f0.y); o0.y = pack2(f0.z, f0.w);
      o0.z = pack2(f1.x, f1.y); o0.w = pack2(f1.z, f1.w);
      o1.x = pack2(f2.x, f2.y); o1.y = pack2(f2.z, f2.w);
      o1.z = pack2(f3.x, f3.y); o1.w = pack2(f3.z, f3.w);
      *(uint4*)(Ash + r * D_MODEL + c0) = o0;
      *(uint4*)(Ash + r * D_MODEL + c0 + 8) = o1;
    }
    __syncthreads();   // before next TDM rewrites Xst
  }

  v8f acc[8];
#pragma unroll
  for (int ns = 0; ns < 8; ++ns)
#pragma unroll
    for (int v = 0; v < 8; ++v) acc[ns][v] = 0.0f;

  const int lb = lane << 4;   // this lane's 16-ushort slot inside a 1KB tile

  for (int fc = 0; fc < 32; ++fc) {
    const int f0 = fc << 7;

    // ---- phase a: G = X@W1, U = X@W3, software-pipelined 1 tile deep -----
    v8f gu[2];
#pragma unroll
    for (int t2 = 0; t2 < 2; ++t2) {
      const unsigned short* tp =
          (t2 ? W3t : W1t) +
          (((size_t)e * 256 + (size_t)(fc * 8 + wvid)) * 32) * 512 + lb;
      v8f c;
#pragma unroll
      for (int v = 0; v < 8; ++v) c[v] = 0.0f;

      // prologue: load tile ks=0
      B256 braw = load2_raw(tp, tp + 8);
      B256 araw = load2_raw(Ash + m * D_MODEL + half * 8,
                            Ash + m * D_MODEL + 16 + half * 8);
      for (int ks = 0; ks < 31; ++ks) {
        // issue loads for ks+1 before consuming ks
        B256 bn = load2_raw(tp + 512, tp + 512 + 8);
        B256 an = load2_raw(Ash + m * D_MODEL + ((ks + 1) << 5) + half * 8,
                            Ash + m * D_MODEL + ((ks + 1) << 5) + 16 + half * 8);
        __builtin_prefetch(tp + 3 * 512, 0, 1);   // 3 K-tiles ahead (L2)
        c = __builtin_amdgcn_wmma_f32_16x16x32_bf16(false, as_bf(araw),
                                                    false, as_bf(braw),
                                                    (short)0, c, false, false);
        braw = bn;
        araw = an;
        tp += 512;
      }
      c = __builtin_amdgcn_wmma_f32_16x16x32_bf16(false, as_bf(araw),
                                                  false, as_bf(braw),
                                                  (short)0, c, false, false);
      gu[t2] = c;
    }

    // bias + SwiGLU -> bf16 H tile in LDS (row-major [16][128])
    // fast silu: x * v_rcp_f32(1+exp(-x)) — result is truncated to bf16
    // anyway, so IEEE-division precision is wasted here.
    {
      const int fcol = f0 + (wvid << 4) + m;
      float bb1 = b1v[e * D_FF + fcol];
      float bb3 = b3v[e * D_FF + fcol];
#pragma unroll
      for (int v = 0; v < 8; ++v) {
        float gg = gu[0][v] + bb1;
        float uu = gu[1][v] + bb3;
        float s  = gg * __builtin_amdgcn_rcpf(1.0f + __expf(-gg));
        Hsh[(v + (half << 3)) * 128 + (wvid << 4) + m] = f2bf(s * uu);
      }
    }
    __syncthreads();

    // ---- phase b: O[:, wvid*128..] += H @ W2[chunk], pipelined -----------
    {
      const size_t w2row = (size_t)e * 64 + (size_t)(wvid * 8);  // ns base
      // tile pointer for flattened iteration it = ns*4 + ks
      #define W2P(itx) (W2t + ((w2row + (size_t)((itx) >> 2)) * 128 + \
                               (size_t)(fc * 4 + ((itx) & 3))) * 512 + lb)
      B256 cur = load2_raw(W2P(0), W2P(0) + 8);
#pragma unroll
      for (int it = 0; it < 32; ++it) {
        B256 nxt = cur;
        if (it < 31) nxt = load2_raw(W2P(it + 1), W2P(it + 1) + 8);
        __builtin_prefetch(W2P(it) + 4 * 512, 0, 1);  // next f-chunk's tile
        v16bf a = as_bf(load2_raw(Hsh + m * 128 + ((it & 3) << 5) + half * 8,
                                  Hsh + m * 128 + ((it & 3) << 5) + 16 + half * 8));
        acc[it >> 2] = __builtin_amdgcn_wmma_f32_16x16x32_bf16(
            false, a, false, as_bf(cur), (short)0, acc[it >> 2], false, false);
        cur = nxt;
      }
      #undef W2P
    }
    __syncthreads();   // all waves done reading H before next chunk
  }

  // ---- epilogue: out[token] += gate_w * (acc + b2), global f32 atomics ---
#pragma unroll
  for (int ns = 0; ns < 8; ++ns) {
    const int n = (wvid << 7) + (ns << 4) + m;
    const float bias = b2v[e * D_MODEL + n];
#pragma unroll
    for (int v = 0; v < 8; ++v) {
      const int M = v + (half << 3);
      if (M < nrow) {
        int   trow = tok[e * CAP_E + row0 + M];
        float gw   = wl[e * CAP_E + row0 + M];
        atomicAdd(out + (size_t)trow * D_MODEL + n, gw * (acc[ns][v] + bias));
      }
    }
  }
}

// ---------------------------------------------------------------------------
extern "C" void kernel_launch(void* const* d_in, const int* in_sizes, int n_in,
                              void* d_out, int out_size, void* d_ws, size_t ws_size,
                              hipStream_t stream) {
  const float* x  = (const float*)d_in[0];
  const float* Wg = (const float*)d_in[1];
  const float* bg = (const float*)d_in[2];
  const float* W1 = (const float*)d_in[3];
  const float* b1 = (const float*)d_in[4];
  const float* W3 = (const float*)d_in[5];
  const float* b3 = (const float*)d_in[6];
  const float* W2 = (const float*)d_in[7];
  const float* b2 = (const float*)d_in[8];
  float* out = (float*)d_out;

  // ws: [cnt|tok|wl] routing state, then bf16 weight images (64 MB each)
  char* ws = (char*)d_ws;
  int*   cnt = (int*)ws;
  int*   tok = (int*)(ws + 64);
  float* wl  = (float*)(ws + 64 + (size_t)NUM_E * CAP_E * sizeof(int));
  const size_t WSZ = (size_t)NUM_E * D_MODEL * D_FF * sizeof(unsigned short); // 64 MB
  unsigned short* W1t = (unsigned short*)(ws + (1u << 20));
  unsigned short* W3t = (unsigned short*)(ws + (1u << 20) + WSZ);
  unsigned short* W2t = (unsigned short*)(ws + (1u << 20) + 2 * WSZ);

  const int n_out = N_TOK * D_MODEL;
  moe_zero<<<(n_out + 255) / 256, 256, 0, stream>>>(out, n_out, cnt);
  moe_gate<<<N_TOK / 8, 256, 0, stream>>>(x, Wg, bg, cnt, tok, wl);

  // 65536 tiles per matrix, 8 tiles (waves) per block
  convert_w<<<8192, 256, 0, stream>>>(W1, W1t,  32, 256, D_FF);
  convert_w<<<8192, 256, 0, stream>>>(W3, W3t,  32, 256, D_FF);
  convert_w<<<8192, 256, 0, stream>>>(W2, W2t, 128,  64, D_MODEL);

  moe_ffn<<<NUM_E * (CAP_E / 16), 256, 0, stream>>>(x, W1t, b1, W3t, b3, W2t, b2,
                                                    cnt, tok, wl, out);
}
